// SelfAttention_Feature_Embedding_Encoder_63256278336192
// MI455X (gfx1250) — matrix operand order repaired and checked by reading, not compiled, not run on hardware
//
#include <hip/hip_runtime.h>
#include <math.h>

// ===================== problem constants =====================
constexpr int BB     = 64;    // B_TRUE * CROPS
constexpr int BTRUE  = 32;
constexpr int DD     = 192;
constexpr int NPATCH = 196;
constexpr int NPAD   = 224;   // padded token count (M side, multiple of 32)
constexpr int KATT   = 256;   // padded key-token count (K side of P@V, multiple of 64)
constexpr int NVALID = 197;
constexpr int NHEAD  = 3;
constexpr int DHEAD  = 64;
constexpr int DFF    = 768;
constexpr int DEPTH  = 12;
constexpr int KIM    = 768;   // 3*16*16 im2col K
constexpr int SATH   = 8;
constexpr int SATDK  = 32;
constexpr int SATC   = 256;   // SATH*SATDK
constexpr int VOC    = 200;
constexpr int MTOK   = BB * NPAD;      // 14336
constexpr int MPATCH = BB * NPATCH;    // 12544
constexpr int MSAT   = MPATCH * 3;     // 37632
constexpr int ZB     = BB * NHEAD;     // 192 attention batches

// ===================== WMMA GEMM =====================
typedef _Float16 v16h __attribute__((ext_vector_type(16)));
typedef float    v8f  __attribute__((ext_vector_type(8)));

#define LDT   72   // LDS tile row stride in halves (144B rows, 16B aligned)
#define KSTEP 64   // K per pipeline stage (two WMMA K-slices)

#define GF_BIAS 1
#define GF_GELU 2
#define GF_RES  4
#define GF_F32  8
#define GF_F16  16

union Frag { v16h v; uint4 q[2]; };

// A fragment 16x32 f16 (ISA 7.12.2): lane m = lane&15;
// lanes 0-15 hold K = {0..7, 16..23}; lanes 16-31 hold K = {8..15, 24..31}.
__device__ __forceinline__ v16h lds_load_a(const _Float16* tile, int rowBase, int lane, int koff) {
  int m  = rowBase + (lane & 15);
  int kb = (lane & 16) ? 8 : 0;
  const _Float16* p = tile + m * LDT + koff;
  Frag f;
  f.q[0] = *(const uint4*)(p + kb);
  f.q[1] = *(const uint4*)(p + 16 + kb);
  return f.v;
}
// B fragment 32x16 f16 fed from B^T tile stored [n][k]:
// lanes 0-15 hold K = {0..7, 16..23} for column n=lane&15; lanes 16-31 hold K = {8..15, 24..31}.
__device__ __forceinline__ v16h lds_load_b(const _Float16* tile, int colBase, int lane, int koff) {
  int n  = colBase + (lane & 15);
  int kb = (lane & 16) ? 8 : 0;
  const _Float16* p = tile + n * LDT + koff + kb;
  Frag f;
  f.q[0] = *(const uint4*)(p);
  f.q[1] = *(const uint4*)(p + 16);
  return f.v;
}

// CDNA5 async global->LDS copy (ASYNCcnt-tracked DMA into LDS, no VGPR staging).
// VDST operand carries the per-lane LDS byte offset; generic LDS pointers encode
// the offset in addr[31:0] (ISA 10.2 aperture mapping).
__device__ __forceinline__ void async_ld128(unsigned lds_off, const void* gptr) {
  asm volatile("global_load_async_to_lds_b128 %0, %1, off"
               :: "v"(lds_off), "v"(gptr)
               : "memory");
}
__device__ __forceinline__ void wait_async0() {
  asm volatile("s_wait_asynccnt 0x0" ::: "memory");
}

// C[M,N](f32 and/or f16) = epilogue( scale * A[M,K] @ Bt[N,K]^T + bias ) (+ residual)
// batched over blockIdx.z with element strides sA/sB/sC. K % 64 == 0.
__global__ __launch_bounds__(128)
void wmma_gemm(const _Float16* __restrict__ A, const _Float16* __restrict__ Bt,
               float* __restrict__ C, _Float16* __restrict__ C16,
               const float* __restrict__ bias, const float* __restrict__ resid,
               int M, int N, int K,
               long sA, long sB, long sC,
               float scale, int flags)
{
  __shared__ alignas(16) _Float16 As[2][64 * LDT];
  __shared__ alignas(16) _Float16 Bs[2][64 * LDT];

  const int z = blockIdx.z;
  A  += (long)z * sA;
  Bt += (long)z * sB;
  const long cbase = (long)z * sC;

  const int tM = blockIdx.y * 64;
  const int tN = blockIdx.x * 64;
  const int tid  = threadIdx.x;
  const int lane = tid & 31;
  const int wave = tid >> 5;
  const int wM = (wave >> 1) * 32;
  const int wN = (wave & 1) * 32;

  // double-buffered async staging of one 64xKSTEP A tile + B^T tile
  auto stage = [&](int buf, int k0) {
    unsigned ab = (unsigned)(size_t)(const void*)&As[buf][0];
    unsigned bb = (unsigned)(size_t)(const void*)&Bs[buf][0];
    #pragma unroll
    for (int c = 0; c < 4; ++c) {
      int idx = tid + c * 128;          // 0..511
      int row = idx >> 3;               // 0..63
      int col = (idx & 7) * 8;          // halves 0..56
      int ar = tM + row; if (ar > M - 1) ar = M - 1;
      int br = tN + row; if (br > N - 1) br = N - 1;
      unsigned loff = (unsigned)((row * LDT + col) * 2);
      async_ld128(ab + loff, (const void*)(A  + (long)ar * K + k0 + col));
      async_ld128(bb + loff, (const void*)(Bt + (long)br * K + k0 + col));
    }
  };

  v8f zero = {0.f,0.f,0.f,0.f,0.f,0.f,0.f,0.f};
  v8f acc[2][2];
  acc[0][0] = zero; acc[0][1] = zero; acc[1][0] = zero; acc[1][1] = zero;

  stage(0, 0);
  wait_async0();
  __syncthreads();

  const int nsteps = K / KSTEP;
  for (int s = 0; s < nsteps; ++s) {
    const int cur = s & 1;
    if (s + 1 < nsteps) stage(cur ^ 1, (s + 1) * KSTEP);   // DMA overlaps WMMA

    const _Float16* at = &As[cur][0];
    const _Float16* bt = &Bs[cur][0];
    v16h alo[2], ahi[2], blo[2], bhi[2];
    alo[0] = lds_load_a(at, wM,      lane, 0);
    alo[1] = lds_load_a(at, wM + 16, lane, 0);
    ahi[0] = lds_load_a(at, wM,      lane, 32);
    ahi[1] = lds_load_a(at, wM + 16, lane, 32);
    blo[0] = lds_load_b(bt, wN,      lane, 0);
    blo[1] = lds_load_b(bt, wN + 16, lane, 0);
    bhi[0] = lds_load_b(bt, wN,      lane, 32);
    bhi[1] = lds_load_b(bt, wN + 16, lane, 32);

    #pragma unroll
    for (int i = 0; i < 2; ++i)
      #pragma unroll
      for (int j = 0; j < 2; ++j) {
        acc[i][j] = __builtin_amdgcn_wmma_f32_16x16x32_f16(false, alo[i], false, blo[j],
                                                           (short)0, acc[i][j], false, false);
        acc[i][j] = __builtin_amdgcn_wmma_f32_16x16x32_f16(false, ahi[i], false, bhi[j],
                                                           (short)0, acc[i][j], false, false);
      }

    if (s + 1 < nsteps) {
      wait_async0();       // next buffer landed
      __syncthreads();     // publish to all waves / all done reading cur
    }
  }

  // C/D f32 layout: VGPR r -> M = r + (lane<16 ? 0 : 8), N = lane&15
  const int hi8 = (lane >> 4) * 8;
  const int n15 = lane & 15;
  #pragma unroll
  for (int i = 0; i < 2; ++i)
    #pragma unroll
    for (int j = 0; j < 2; ++j)
      #pragma unroll
      for (int r = 0; r < 8; ++r) {
        int m = tM + wM + i * 16 + r + hi8;
        int n = tN + wN + j * 16 + n15;
        if (m < M && n < N) {
          float v = acc[i][j][r] * scale;
          if (flags & GF_BIAS) v += bias[n];
          if (flags & GF_GELU) v = 0.5f * v * (1.0f + erff(v * 0.70710678118f));
          long o = cbase + (long)m * N + n;
          if (flags & GF_RES)  v += resid[o];
          if (flags & GF_F32)  C[o]   = v;
          if (flags & GF_F16)  C16[o] = (_Float16)v;
        }
      }
}

// ===================== elementwise / glue kernels =====================
__global__ void cast16_k(const float* __restrict__ s, _Float16* __restrict__ d, long n) {
  long i = (long)blockIdx.x * blockDim.x + threadIdx.x;
  if (i < n) d[i] = (_Float16)s[i];
}

// (L,K,N) f32 -> (L,N,K) f16
__global__ void tcast_k(const float* __restrict__ s, _Float16* __restrict__ d,
                        int L, int Kd, int Nd, long total) {
  long i = (long)blockIdx.x * blockDim.x + threadIdx.x;
  if (i >= total) return;
  int n = (int)(i % Nd);
  long rest = i / Nd;
  int k = (int)(rest % Kd);
  int l = (int)(rest / Kd);
  d[((long)l * Nd + n) * Kd + k] = (_Float16)s[i];
}

// sift_em[b,d] = sum_k (hist[b,k]/sum(hist[b,:])) * emb[k,d]
__global__ void sift_k(const float* __restrict__ hist, const float* __restrict__ emb,
                       float* __restrict__ out) {
  int i = blockIdx.x * blockDim.x + threadIdx.x;
  if (i >= BB * DD) return;
  int d = i % DD, b = i / DD;
  const float* hb = hist + (long)b * VOC;
  float s = 0.f;
  for (int k = 0; k < VOC; ++k) s += hb[k];
  float inv = (s != 0.f && s == s) ? 1.f / s : 0.f;
  float acc = 0.f;
  for (int k = 0; k < VOC; ++k) acc += hb[k] * emb[(long)k * DD + d];
  out[i] = acc * inv;
}

// x(B,3,224,224) -> im2col (B*196, 768) f16 ; col = c*256 + ph*16 + pw
__global__ void im2col_k(const float* __restrict__ x, _Float16* __restrict__ A, long total) {
  long i = (long)blockIdx.x * blockDim.x + threadIdx.x;
  if (i >= total) return;
  int col = (int)(i % KIM);
  long row = i / KIM;
  int b = (int)(row / NPATCH);
  int p = (int)(row % NPATCH);
  int c  = col >> 8;
  int ph = (col >> 4) & 15;
  int pw = col & 15;
  int py = p / 14, px = p % 14;
  A[i] = (_Float16)x[((long)(b * 3 + c) * 224 + (py * 16 + ph)) * 224 + px * 16 + pw];
}

// at_in rows (bp*3+s) x 192 f16
__global__ void build_at_in_k(const float* __restrict__ tpatch, const int* __restrict__ eidx,
                              const float* __restrict__ eleva, const float* __restrict__ sift_em,
                              _Float16* __restrict__ A, long total) {
  long i = (long)blockIdx.x * blockDim.x + threadIdx.x;
  if (i >= total) return;
  int d = (int)(i % DD);
  long r = i / DD;
  int s = (int)(r % 3);
  long bp = r / 3;
  int b = (int)(bp / NPATCH);
  float v;
  if (s == 0)      v = tpatch[bp * DD + d];
  else if (s == 1) v = eleva[(long)eidx[b & (BTRUE - 1)] * DD + d];
  else             v = sift_em[(long)(b & (BTRUE - 1)) * DD + d];
  A[i] = (_Float16)v;
}

// tiny 3-token attention per (bp, head, query-slot); SATDK=32 dims
__global__ void sat_attn_k(const _Float16* __restrict__ Q, const _Float16* __restrict__ K,
                           const _Float16* __restrict__ V, _Float16* __restrict__ O) {
  long gid = (long)blockIdx.x * blockDim.x + threadIdx.x;
  if (gid >= (long)MPATCH * SATH * 3) return;
  int qs = (int)(gid % 3);
  int hh = (int)((gid / 3) % SATH);
  long bp = gid / (3 * SATH);
  long base = bp * 3 * SATC + hh * SATDK;
  float q[SATDK];
  #pragma unroll
  for (int d = 0; d < SATDK; ++d) q[d] = (float)Q[base + (long)qs * SATC + d];
  float dots[3];
  float mx = -1e30f;
  for (int ks = 0; ks < 3; ++ks) {
    float acc = 0.f;
    #pragma unroll
    for (int d = 0; d < SATDK; ++d) acc += q[d] * (float)K[base + (long)ks * SATC + d];
    dots[ks] = acc * 0.17677669529f;  // 1/sqrt(32)
    mx = fmaxf(mx, dots[ks]);
  }
  float e0 = __expf(dots[0] - mx), e1 = __expf(dots[1] - mx), e2 = __expf(dots[2] - mx);
  float inv = 1.f / (e0 + e1 + e2);
  #pragma unroll
  for (int d = 0; d < SATDK; ++d) {
    float o = (e0 * (float)V[base + 0 * SATC + d] +
               e1 * (float)V[base + 1 * SATC + d] +
               e2 * (float)V[base + 2 * SATC + d]) * inv;
    O[base + (long)qs * SATC + d] = (_Float16)o;
  }
}

// h(B,224,192) f32: n==0 -> cls+pos0 ; 1..196 -> sum_s oproj + pos[n] ; pad -> 0
__global__ void build_h_k(const float* __restrict__ oproj, const float* __restrict__ cls,
                          const float* __restrict__ pos, float* __restrict__ H, long total) {
  long i = (long)blockIdx.x * blockDim.x + threadIdx.x;
  if (i >= total) return;
  int d = (int)(i % DD);
  long r = i / DD;
  int n = (int)(r % NPAD);
  int b = (int)(r / NPAD);
  float v = 0.f;
  if (n == 0) {
    v = cls[d] + pos[d];
  } else if (n <= NPATCH) {
    long bp = (long)b * NPATCH + (n - 1);
    v = oproj[(bp * 3 + 0) * DD + d] + oproj[(bp * 3 + 1) * DD + d] +
        oproj[(bp * 3 + 2) * DD + d] + pos[(long)n * DD + d];
  }
  H[i] = v;
}

__device__ __forceinline__ float wsum(float v) {
  #pragma unroll
  for (int o = 16; o > 0; o >>= 1) v += __shfl_xor(v, o, 32);
  return v;
}
__device__ __forceinline__ float wmax(float v) {
  #pragma unroll
  for (int o = 16; o > 0; o >>= 1) v = fmaxf(v, __shfl_xor(v, o, 32));
  return v;
}

// one wave32 per row, D=192
__global__ void layernorm16_k(const float* __restrict__ X, const float* __restrict__ w,
                              const float* __restrict__ b, _Float16* __restrict__ Y) {
  long row = blockIdx.x;
  int t = threadIdx.x;
  const float* x = X + row * DD;
  float s = 0.f, s2 = 0.f;
  for (int c = t; c < DD; c += 32) { float v = x[c]; s += v; s2 += v * v; }
  s = wsum(s); s2 = wsum(s2);
  float mu = s * (1.0f / DD);
  float var = s2 * (1.0f / DD) - mu * mu;
  float rinv = rsqrtf(var + 1e-5f);
  for (int c = t; c < DD; c += 32)
    Y[row * DD + c] = (_Float16)((x[c] - mu) * rinv * w[c] + b[c]);
}

// final LN over token 0 of each batch row; f32 output (B,192)
__global__ void ln_final_k(const float* __restrict__ H, const float* __restrict__ w,
                           const float* __restrict__ b, float* __restrict__ out) {
  int bi = blockIdx.x;
  int t = threadIdx.x;
  const float* x = H + (long)bi * NPAD * DD;  // token 0 row
  float s = 0.f, s2 = 0.f;
  for (int c = t; c < DD; c += 32) { float v = x[c]; s += v; s2 += v * v; }
  s = wsum(s); s2 = wsum(s2);
  float mu = s * (1.0f / DD);
  float var = s2 * (1.0f / DD) - mu * mu;
  float rinv = rsqrtf(var + 1e-5f);
  for (int c = t; c < DD; c += 32)
    out[(long)bi * DD + c] = (x[c] - mu) * rinv * w[c] + b[c];
}

// qkv16 (B*224, 576) -> per-(b,h): Q,K (224x64); V^T (64x256) zero-padded past 224
__global__ void split_qkv_k(const _Float16* __restrict__ QKV, _Float16* __restrict__ Q,
                            _Float16* __restrict__ K, _Float16* __restrict__ VT, long total) {
  long i = (long)blockIdx.x * blockDim.x + threadIdx.x;
  if (i >= total) return;
  int d = (int)(i % DHEAD);
  int h = (int)((i / DHEAD) % NHEAD);
  int n = (int)((i / ((long)DHEAD * NHEAD)) % KATT);
  int b = (int)(i / ((long)DHEAD * NHEAD * KATT));
  int z = b * NHEAD + h;
  if (n < NPAD) {
    long src = ((long)b * NPAD + n) * (3 * DD);
    Q [((long)z * NPAD + n) * DHEAD + d] = QKV[src + 0 * DD + h * DHEAD + d];
    K [((long)z * NPAD + n) * DHEAD + d] = QKV[src + 1 * DD + h * DHEAD + d];
    VT[((long)z * DHEAD + d) * KATT + n] = QKV[src + 2 * DD + h * DHEAD + d];
  } else {
    VT[((long)z * DHEAD + d) * KATT + n] = (_Float16)0.f;  // K-pad rows must be exact zero
  }
}

// one wave per row of (ZB*224) score rows; scores have 224 cols, probs 256 cols.
// cols >= NVALID get probability exactly 0 (also zeroing the K-padding 224..255).
__global__ void softmax_rows_k(const float* __restrict__ S, _Float16* __restrict__ P) {
  long row = blockIdx.x;
  int t = threadIdx.x;
  const float* s = S + row * NPAD;
  _Float16* p = P + row * KATT;
  float mx = -1e30f;
  for (int c = t; c < NVALID; c += 32) mx = fmaxf(mx, s[c]);
  mx = wmax(mx);
  float sum = 0.f;
  for (int c = t; c < NVALID; c += 32) sum += __expf(s[c] - mx);
  sum = wsum(sum);
  float inv = 1.f / sum;
  for (int c = t; c < KATT; c += 32)
    p[c] = (c < NVALID) ? (_Float16)(__expf(s[c] - mx) * inv) : (_Float16)0.f;
}

// ctxz (z=b*3+h, 224, 64) f16 -> ctx16 (B*224, 192) f16 with col = h*64+d
__global__ void gather_ctx_k(const _Float16* __restrict__ CZ, _Float16* __restrict__ C, long total) {
  long i = (long)blockIdx.x * blockDim.x + threadIdx.x;
  if (i >= total) return;
  int col = (int)(i % DD);
  int h = col / DHEAD, d = col % DHEAD;
  int n = (int)((i / DD) % NPAD);
  int b = (int)(i / ((long)DD * NPAD));
  C[i] = CZ[(((long)(b * NHEAD + h) * NPAD) + n) * DHEAD + d];
}

// ===================== host orchestration =====================
static inline void launch_gemm(hipStream_t st, const _Float16* A, const _Float16* Bt,
                               float* C, _Float16* C16, const float* bias, const float* resid,
                               int M, int N, int K, long sA, long sB, long sC, int nz,
                               float scale, int flags) {
  dim3 g((N + 63) / 64, (M + 63) / 64, nz);
  wmma_gemm<<<g, 128, 0, st>>>(A, Bt, C, C16, bias, resid, M, N, K, sA, sB, sC, scale, flags);
}

static inline unsigned nblk(long n) { return (unsigned)((n + 255) / 256); }

extern "C" void kernel_launch(void* const* d_in, const int* in_sizes, int n_in,
                              void* d_out, int out_size, void* d_ws, size_t ws_size,
                              hipStream_t stream) {
  (void)in_sizes; (void)n_in; (void)out_size; (void)ws_size;
  const float* x         = (const float*)d_in[0];
  const int*   ele_idx   = (const int*)  d_in[1];
  const float* sift_hist = (const float*)d_in[2];
  const float* patch_w   = (const float*)d_in[3];
  const float* patch_b   = (const float*)d_in[4];
  const float* cls_tok   = (const float*)d_in[5];
  const float* pos_emb   = (const float*)d_in[6];
  const float* eleva     = (const float*)d_in[7];
  const float* sift_emb  = (const float*)d_in[8];
  const float* sat_wq    = (const float*)d_in[9];
  const float* sat_bq    = (const float*)d_in[10];
  const float* sat_wk    = (const float*)d_in[11];
  const float* sat_bk    = (const float*)d_in[12];
  const float* sat_wv    = (const float*)d_in[13];
  const float* sat_bv    = (const float*)d_in[14];
  const float* sat_wo    = (const float*)d_in[15];
  const float* sat_bo    = (const float*)d_in[16];
  const float* ln1_w     = (const float*)d_in[17];
  const float* ln1_b     = (const float*)d_in[18];
  const float* qkv_w     = (const float*)d_in[19];
  const float* proj_w    = (const float*)d_in[20];
  const float* proj_b    = (const float*)d_in[21];
  const float* ln2_w     = (const float*)d_in[22];
  const float* ln2_b     = (const float*)d_in[23];
  const float* fc1_w     = (const float*)d_in[24];
  const float* fc1_b     = (const float*)d_in[25];
  const float* fc2_w     = (const float*)d_in[26];
  const float* fc2_b     = (const float*)d_in[27];
  const float* norm_w    = (const float*)d_in[28];
  const float* norm_b    = (const float*)d_in[29];
  float* out = (float*)d_out;

  char* ws = (char*)d_ws;
  size_t off = 0;
  auto alloc = [&](size_t bytes) { size_t o = off; off += (bytes + 255) & ~(size_t)255; return o; };

  // ---- persistent ----
  float*    h        = (float*)   (ws + alloc((size_t)MTOK * DD * 4));
  float*    sift_em  = (float*)   (ws + alloc((size_t)BB * DD * 4));
  _Float16* wt_patch = (_Float16*)(ws + alloc((size_t)DD * KIM * 2));
  _Float16* wt_sq    = (_Float16*)(ws + alloc((size_t)SATC * DD * 2));
  _Float16* wt_sk    = (_Float16*)(ws + alloc((size_t)SATC * DD * 2));
  _Float16* wt_sv    = (_Float16*)(ws + alloc((size_t)SATC * DD * 2));
  _Float16* wt_so    = (_Float16*)(ws + alloc((size_t)DD * SATC * 2));
  _Float16* wt_qkv   = (_Float16*)(ws + alloc((size_t)DEPTH * 3 * DD * DD * 2));
  _Float16* wt_proj  = (_Float16*)(ws + alloc((size_t)DEPTH * DD * DD * 2));
  _Float16* wt_fc1   = (_Float16*)(ws + alloc((size_t)DEPTH * DFF * DD * 2));
  _Float16* wt_fc2   = (_Float16*)(ws + alloc((size_t)DEPTH * DD * DFF * 2));
  size_t scratch0 = off;

  // ---- stage-0 scratch (dead after build_h) ----
  _Float16* a_im    = (_Float16*)(ws + alloc((size_t)MPATCH * KIM * 2));
  float*    t_patch = (float*)   (ws + alloc((size_t)MPATCH * DD * 4));
  _Float16* at_in   = (_Float16*)(ws + alloc((size_t)MSAT * DD * 2));
  _Float16* qsat    = (_Float16*)(ws + alloc((size_t)MSAT * SATC * 2));
  _Float16* ksat    = (_Float16*)(ws + alloc((size_t)MSAT * SATC * 2));
  _Float16* vsat    = (_Float16*)(ws + alloc((size_t)MSAT * SATC * 2));
  _Float16* osat    = (_Float16*)(ws + alloc((size_t)MSAT * SATC * 2));
  float*    oproj   = (float*)   (ws + alloc((size_t)MSAT * DD * 4));

  // ---- main-stage scratch (overlaps stage-0 region) ----
  off = scratch0;
  _Float16* y16    = (_Float16*)(ws + alloc((size_t)MTOK * DD * 2));
  _Float16* qkv16  = (_Float16*)(ws + alloc((size_t)MTOK * 3 * DD * 2));
  _Float16* qh     = (_Float16*)(ws + alloc((size_t)ZB * NPAD * DHEAD * 2));
  _Float16* kh     = (_Float16*)(ws + alloc((size_t)ZB * NPAD * DHEAD * 2));
  _Float16* vT     = (_Float16*)(ws + alloc((size_t)ZB * DHEAD * KATT * 2));
  float*    scores = (float*)   (ws + alloc((size_t)ZB * NPAD * NPAD * 4));
  _Float16* probs  = (_Float16*)(ws + alloc((size_t)ZB * NPAD * KATT * 2));
  _Float16* ctxz   = (_Float16*)(ws + alloc((size_t)ZB * NPAD * DHEAD * 2));
  _Float16* ctx16  = (_Float16*)(ws + alloc((size_t)MTOK * DD * 2));
  _Float16* hg16   = (_Float16*)(ws + alloc((size_t)MTOK * DFF * 2));

  // ================= weight precast (f16, B^T layout [N][K]) =================
  cast16_k<<<nblk((long)DD * KIM), 256, 0, stream>>>(patch_w, wt_patch, (long)DD * KIM);
  tcast_k<<<nblk((long)DD * SATC), 256, 0, stream>>>(sat_wq, wt_sq, 1, DD, SATC, (long)DD * SATC);
  tcast_k<<<nblk((long)DD * SATC), 256, 0, stream>>>(sat_wk, wt_sk, 1, DD, SATC, (long)DD * SATC);
  tcast_k<<<nblk((long)DD * SATC), 256, 0, stream>>>(sat_wv, wt_sv, 1, DD, SATC, (long)DD * SATC);
  tcast_k<<<nblk((long)SATC * DD), 256, 0, stream>>>(sat_wo, wt_so, 1, SATC, DD, (long)SATC * DD);
  tcast_k<<<nblk((long)DEPTH * DD * 3 * DD), 256, 0, stream>>>(qkv_w,  wt_qkv,  DEPTH, DD, 3 * DD, (long)DEPTH * DD * 3 * DD);
  tcast_k<<<nblk((long)DEPTH * DD * DD),     256, 0, stream>>>(proj_w, wt_proj, DEPTH, DD, DD,     (long)DEPTH * DD * DD);
  tcast_k<<<nblk((long)DEPTH * DD * DFF),    256, 0, stream>>>(fc1_w,  wt_fc1,  DEPTH, DD, DFF,    (long)DEPTH * DD * DFF);
  tcast_k<<<nblk((long)DEPTH * DFF * DD),    256, 0, stream>>>(fc2_w,  wt_fc2,  DEPTH, DFF, DD,    (long)DEPTH * DFF * DD);

  // ================= stage 0: embeddings + SAT attention =================
  sift_k<<<nblk((long)BB * DD), 256, 0, stream>>>(sift_hist, sift_emb, sift_em);
  im2col_k<<<nblk((long)MPATCH * KIM), 256, 0, stream>>>(x, a_im, (long)MPATCH * KIM);
  launch_gemm(stream, a_im, wt_patch, t_patch, nullptr, patch_b, nullptr,
              MPATCH, DD, KIM, 0, 0, 0, 1, 1.f, GF_BIAS | GF_F32);
  build_at_in_k<<<nblk((long)MSAT * DD), 256, 0, stream>>>(t_patch, ele_idx, eleva, sift_em,
                                                           at_in, (long)MSAT * DD);
  launch_gemm(stream, at_in, wt_sq, nullptr, qsat, sat_bq, nullptr,
              MSAT, SATC, DD, 0, 0, 0, 1, 1.f, GF_BIAS | GF_F16);
  launch_gemm(stream, at_in, wt_sk, nullptr, ksat, sat_bk, nullptr,
              MSAT, SATC, DD, 0, 0, 0, 1, 1.f, GF_BIAS | GF_F16);
  launch_gemm(stream, at_in, wt_sv, nullptr, vsat, sat_bv, nullptr,
              MSAT, SATC, DD, 0, 0, 0, 1, 1.f, GF_BIAS | GF_F16);
  sat_attn_k<<<nblk((long)MPATCH * SATH * 3), 256, 0, stream>>>(qsat, ksat, vsat, osat);
  launch_gemm(stream, osat, wt_so, oproj, nullptr, sat_bo, nullptr,
              MSAT, DD, SATC, 0, 0, 0, 1, 1.f, GF_BIAS | GF_F32);
  build_h_k<<<nblk((long)MTOK * DD), 256, 0, stream>>>(oproj, cls_tok, pos_emb, h, (long)MTOK * DD);

  // ================= transformer blocks =================
  for (int l = 0; l < DEPTH; ++l) {
    layernorm16_k<<<MTOK, 32, 0, stream>>>(h, ln1_w + (long)l * DD, ln1_b + (long)l * DD, y16);
    launch_gemm(stream, y16, wt_qkv + (long)l * 3 * DD * DD, nullptr, qkv16, nullptr, nullptr,
                MTOK, 3 * DD, DD, 0, 0, 0, 1, 1.f, GF_F16);
    split_qkv_k<<<nblk((long)BB * KATT * NHEAD * DHEAD), 256, 0, stream>>>(
        qkv16, qh, kh, vT, (long)BB * KATT * NHEAD * DHEAD);
    launch_gemm(stream, qh, kh, scores, nullptr, nullptr, nullptr,
                NPAD, NPAD, DHEAD,
                (long)NPAD * DHEAD, (long)NPAD * DHEAD, (long)NPAD * NPAD, ZB,
                0.125f, GF_F32);  // scale = (D/H)^-0.5 = 1/8
    softmax_rows_k<<<ZB * NPAD, 32, 0, stream>>>(scores, probs);
    launch_gemm(stream, probs, vT, nullptr, ctxz, nullptr, nullptr,
                NPAD, DHEAD, KATT,
                (long)NPAD * KATT, (long)DHEAD * KATT, (long)NPAD * DHEAD, ZB,
                1.f, GF_F16);
    gather_ctx_k<<<nblk((long)MTOK * DD), 256, 0, stream>>>(ctxz, ctx16, (long)MTOK * DD);
    launch_gemm(stream, ctx16, wt_proj + (long)l * DD * DD, h, nullptr,
                proj_b + (long)l * DD, h,
                MTOK, DD, DD, 0, 0, 0, 1, 1.f, GF_BIAS | GF_RES | GF_F32);
    layernorm16_k<<<MTOK, 32, 0, stream>>>(h, ln2_w + (long)l * DD, ln2_b + (long)l * DD, y16);
    launch_gemm(stream, y16, wt_fc1 + (long)l * DFF * DD, nullptr, hg16,
                fc1_b + (long)l * DFF, nullptr,
                MTOK, DFF, DD, 0, 0, 0, 1, 1.f, GF_BIAS | GF_GELU | GF_F16);
    launch_gemm(stream, hg16, wt_fc2 + (long)l * DD * DFF, h, nullptr,
                fc2_b + (long)l * DD, h,
                MTOK, DD, DFF, 0, 0, 0, 1, 1.f, GF_BIAS | GF_RES | GF_F32);
  }

  // ================= final LN, token 0 only =================
  ln_final_k<<<BB, 32, 0, stream>>>(h, norm_w, norm_b, out);
}
